// MI_Module_3917010174657
// MI455X (gfx1250) — matrix operand order repaired and checked by reading, not compiled, not run on hardware
//
#include <hip/hip_runtime.h>

typedef __attribute__((ext_vector_type(16))) __bf16 v16bf;
typedef __attribute__((ext_vector_type(8)))  __bf16 v8bf;   // 16 B, align 16
typedef __attribute__((ext_vector_type(8)))  float  v8f;

#define Bn 8
#define Cn 128
#define Hn 96
#define Wn 96
#define HWn (Hn * Wn)        // 9216
#define Pn  (Bn * HWn)       // 73728 spatial points

#define ROWP 72              // padded LDS row (bf16 units): 144 B, 36-word stride

// ---------------------------------------------------------------------------
// Kernel 0: repack Fw (128x128 f32) into bf16 WMMA A-fragment order.
//   Afrag[((mt*4 + kt)*32 + lane)*16 + e]
//   M = mt*16 + (lane & 15),  K = kt*32 + (e>>3)*16 + (lane>>4)*8 + (e&7)
// ---------------------------------------------------------------------------
__global__ void pack_fw_kernel(const float* __restrict__ Fw,
                               __bf16* __restrict__ Afrag) {
    for (int idx = threadIdx.x; idx < Cn * Cn; idx += blockDim.x) {
        int e    = idx & 15;
        int lane = (idx >> 4) & 31;
        int kt   = (idx >> 9) & 3;
        int mt   = idx >> 11;
        int M    = mt * 16 + (lane & 15);
        int K    = kt * 32 + (e >> 3) * 16 + (lane >> 4) * 8 + (e & 7);
        Afrag[idx] = (__bf16)Fw[M * Cn + K];
    }
}

// ---------------------------------------------------------------------------
// Kernel 1: fused multi-dilation depthwise 3x3 + BN/ReLU + per-channel 4->1
// mix + BN/ReLU. Thread per (c, p), p fastest: coalesced taps and stores.
// y written bf16 in [C][P] layout (rows are WMMA B-matrix feeds).
// ---------------------------------------------------------------------------
__global__ void dwpw_kernel(const float* __restrict__ x,
                            const float* __restrict__ Dw,
                            const float* __restrict__ Dscale,
                            const float* __restrict__ Dbias,
                            const float* __restrict__ Pw,
                            const float* __restrict__ Pscale,
                            const float* __restrict__ Pbias,
                            __bf16* __restrict__ y) {
    const int c  = blockIdx.y;
    const int p  = blockIdx.x * blockDim.x + threadIdx.x;
    const int b  = p / HWn;
    const int hw = p % HWn;
    const int h  = hw / Wn;
    const int w  = hw % Wn;
    const float* __restrict__ xc = x + (b * Cn + c) * HWn;

    float s = 0.0f;
#pragma unroll
    for (int i = 0; i < 4; ++i) {
        const int d = 1 << i;                         // dilation 1,2,4,8
        const float* __restrict__ wk = Dw + (i * Cn + c) * 9;
        float acc = 0.0f;
#pragma unroll
        for (int kh = 0; kh < 3; ++kh) {
#pragma unroll
            for (int kw = 0; kw < 3; ++kw) {
                const int hh = h + (kh - 1) * d;
                const int ww = w + (kw - 1) * d;
                float v = 0.0f;
                if (hh >= 0 && hh < Hn && ww >= 0 && ww < Wn)
                    v = xc[hh * Wn + ww];             // zero padding
                acc = fmaf(v, wk[kh * 3 + kw], acc);
            }
        }
        const float m = fmaxf(fmaf(acc, Dscale[i * Cn + c], Dbias[i * Cn + c]), 0.0f);
        s = fmaf(Pw[c * 4 + i], m, s);
    }
    const float yv = fmaxf(fmaf(s, Pscale[c], Pbias[c]), 0.0f);
    y[c * Pn + p] = (__bf16)yv;
}

// ---------------------------------------------------------------------------
// Kernel 2: 1x1 conv C->C as bf16 WMMA GEMM with async-to-LDS double-buffered
// B staging. Block = 256 threads (8 waves) owns 128(M) x 64(N); the shared
// 32(K) x 64(N) bf16 B sub-tile is DMA'd to LDS ONCE per block per K-step
// (global_load_async_to_lds_b128, ASYNCcnt), killing the 8x redundant L2
// reads of y and overlapping the next stage's DMA with current WMMAs.
// ---------------------------------------------------------------------------
__global__ void gemm_fuse_kernel(const __bf16* __restrict__ y,
                                 const __bf16* __restrict__ Afrag,
                                 const float* __restrict__ x,
                                 const float* __restrict__ Fscale,
                                 const float* __restrict__ Fbias,
                                 float* __restrict__ out) {
    __shared__ __align__(32) __bf16 ytile[2][32 * ROWP];   // 2 x 4.5 KB

    const int p0   = blockIdx.x * 64;       // pixel tile base (never crosses b)
    const int b    = p0 / HWn;
    const int hw0  = p0 % HWn;
    const int tid  = threadIdx.x;
    const int lane = tid & 31;
    const int mt   = tid >> 5;              // wave id 0..7 -> M-tile

    // Per-thread async copy slot: 16 B (8 bf16). row = tid/8, col = (tid%8)*8.
    const int srow = tid >> 3;              // 0..31 (K row within stage)
    const int scol = (tid & 7) * 8;         // 0..56
    const unsigned long long gbase = (unsigned long long)(const void*)(y + p0);

    // A fragments for this wave's 16 M rows, all 4 K-steps (reused 4x each).
    v16bf a[4];
#pragma unroll
    for (int kt = 0; kt < 4; ++kt)
        a[kt] = *(const v16bf*)(Afrag + ((mt * 4 + kt) * 32 + lane) * 16);

    // Epilogue constants: D layout => VGPR r holds M = mt*16 + (lane>>4)*8 + r.
    const int mrowbase = mt * 16 + (lane >> 4) * 8;
    float fs[8], fb[8];
#pragma unroll
    for (int r = 0; r < 8; ++r) {
        fs[r] = Fscale[mrowbase + r];
        fb[r] = Fbias[mrowbase + r];
    }
    const int ncol = lane & 15;             // D layout: lane%16 = N column
    __builtin_prefetch(x + (b * Cn + mrowbase) * HWn + hw0, 0, 0);

    // ---- async DMA of one 32x64 bf16 stage into LDS buffer `sel` ----------
    auto issue_stage = [&](int kt, int sel) {
        // flat LDS address low 32 bits == LDS byte offset (ISA 10.2 aperture)
        const unsigned lds_off =
            (unsigned)(unsigned long long)(const void*)&ytile[sel][srow * ROWP + scol];
        const unsigned goff = (unsigned)(((kt * 32 + srow) * Pn + scol) * 2);
        asm volatile("global_load_async_to_lds_b128 %0, %1, %2"
                     :: "v"(lds_off), "v"(goff), "s"(gbase)
                     : "memory");
    };

    issue_stage(0, 0);                                    // prologue

    v8f acc[4];
#pragma unroll
    for (int nt = 0; nt < 4; ++nt) acc[nt] = (v8f){};

#pragma unroll
    for (int kt = 0; kt < 4; ++kt) {
        if (kt < 3) {
            issue_stage(kt + 1, (kt + 1) & 1);            // prefetch next stage
            asm volatile("s_wait_asynccnt 0x1" ::: "memory");
        } else {
            asm volatile("s_wait_asynccnt 0x0" ::: "memory");
        }
        __syncthreads();                                   // stage kt visible

        const int sel = kt & 1;
        const __bf16* brow = &ytile[sel][lane & 15];       // B: lane%16 = N column
#pragma unroll
        for (int nt = 0; nt < 4; ++nt) {
            const int n0 = nt * 16;
            v16bf bf;
#pragma unroll
            for (int s = 0; s < 16; ++s)
                bf[s] = brow[((s >> 3) * 16 + (lane >> 4) * 8 + (s & 7)) * ROWP + n0];
            acc[nt] = __builtin_amdgcn_wmma_f32_16x16x32_bf16(
                false, a[kt], false, bf, (short)0, acc[nt], false, false);
        }
        __syncthreads();                                   // buffer reuse fence
    }

    // Fused epilogue: affine + ReLU + residual + ReLU, coalesced stores.
#pragma unroll
    for (int nt = 0; nt < 4; ++nt) {
        const int n0 = nt * 16;
#pragma unroll
        for (int r = 0; r < 8; ++r) {
            const int m    = mrowbase + r;
            const int oidx = (b * Cn + m) * HWn + hw0 + n0 + ncol;
            float v = fmaf(acc[nt][r], fs[r], fb[r]);
            v = fmaxf(v, 0.0f);
            v = fmaxf(v + x[oidx], 0.0f);
            out[oidx] = v;
        }
    }
}

// ---------------------------------------------------------------------------
// Launch. Inputs: 0:x 1:Dw 2:Dscale 3:Dbias 4:Pw 5:Pscale 6:Pbias 7:Fw
//                 8:Fscale 9:Fbias
// Workspace: [0, 32KB) = Afrag bf16; [32KB, ~18.9MB) = y bf16 [C][P].
// ---------------------------------------------------------------------------
extern "C" void kernel_launch(void* const* d_in, const int* in_sizes, int n_in,
                              void* d_out, int out_size, void* d_ws, size_t ws_size,
                              hipStream_t stream) {
    const float* x      = (const float*)d_in[0];
    const float* Dw     = (const float*)d_in[1];
    const float* Dscale = (const float*)d_in[2];
    const float* Dbias  = (const float*)d_in[3];
    const float* Pw     = (const float*)d_in[4];
    const float* Pscale = (const float*)d_in[5];
    const float* Pbias  = (const float*)d_in[6];
    const float* Fw     = (const float*)d_in[7];
    const float* Fscale = (const float*)d_in[8];
    const float* Fbias  = (const float*)d_in[9];
    float* out = (float*)d_out;

    __bf16* Afrag = (__bf16*)d_ws;                          // 32 KB
    __bf16* yws   = (__bf16*)((char*)d_ws + 32768);         // 18.87 MB

    pack_fw_kernel<<<1, 256, 0, stream>>>(Fw, Afrag);
    dwpw_kernel<<<dim3(Pn / 256, Cn), 256, 0, stream>>>(
        x, Dw, Dscale, Dbias, Pw, Pscale, Pbias, yws);
    gemm_fuse_kernel<<<Pn / 64, 256, 0, stream>>>(
        yws, Afrag, x, Fscale, Fbias, out);
}
